// MsGroupConv1d_15436112462318
// MI455X (gfx1250) — matrix operand order repaired and checked by reading, not compiled, not run on hardware
//
#include <hip/hip_runtime.h>
#include <hip/hip_bf16.h>

typedef __bf16 bf16_t;
typedef bf16_t v16bf __attribute__((ext_vector_type(16)));
typedef bf16_t v8bf  __attribute__((ext_vector_type(8)));
typedef float  v8f   __attribute__((ext_vector_type(8)));

namespace {
constexpr int kC       = 256;          // channels (M and K of the GEMM)
constexpr int kHW      = 56 * 56;      // 3136 spatial positions per image
constexpr int kT       = 8;            // temporal segment length
constexpr int kNT      = 64;           // nt = n_batch * n_segment
constexpr int kNTile   = 64;           // spatial positions per workgroup
constexpr int kKStep   = 32;           // K per WMMA step (bf16 16x16x32)
constexpr int kKSteps  = kC / kKStep;  // 8
constexpr int kYStride = 40;           // padded bf16 row stride of staged B tile
constexpr int kWElems  = kC * kC;      // 65536 bf16 = 128 KB weight in LDS
constexpr int kYElems  = kNTile * kYStride;  // 2560 bf16 = 5 KB per Y buffer
constexpr int kLdsBytes = (kWElems + 2 * kYElems) * 2;
} // namespace

// Pre-pass: convert net_w (256x256 f32, o-major / k-contiguous) to bf16 scratch.
__global__ void msconv_wcvt_kernel(const float* __restrict__ w,
                                   bf16_t* __restrict__ wb) {
    int i = blockIdx.x * blockDim.x + threadIdx.x;
    if (i < kC * kC) wb[i] = (bf16_t)w[i];
}

// Fused temporal-shift-mask + 1x1-conv GEMM, double-buffered B tile.
// grid = (kHW/kNTile = 49, kNT = 64), block = 256 threads (8 waves).
__global__ __launch_bounds__(256)
void msconv_gemm_kernel(const float* __restrict__ x,
                        const bf16_t* __restrict__ wb,
                        float* __restrict__ out) {
    // Single shared base, all accesses via index arithmetic so the compiler
    // keeps everything in addrspace(3) -> ds_load/ds_store (no flat/global).
    extern __shared__ bf16_t lds[];  // [0,kWElems): W ; then 2 Y buffers

    const int tid    = threadIdx.x;
    const int lane   = tid & 31;
    const int waveid = __builtin_amdgcn_readfirstlane(tid >> 5);  // scalar wave id
    const int img    = blockIdx.y;          // nt index in [0,64)
    const int nb     = img >> 3;            // batch
    const int tt     = img & 7;             // segment position
    const int p0     = blockIdx.x * kNTile; // spatial tile base

    // Stage the full bf16 weight matrix into LDS once (128 KB of 320 KB WGP LDS).
    {
        const uint4* src = (const uint4*)wb;
        uint4*       dst = (uint4*)lds;
        for (int j = tid; j < (kWElems * 2) / 16; j += 256) dst[j] = src[j];
    }

    const int plocal = tid & 63;            // 0..63 spatial within tile
    const int pg     = p0 + plocal;         // global spatial position
    const int k8     = (waveid >> 1) * 8;   // scalar: thread-group's 8 channels

    // Shifted-sum builder for channels [kc+k8, kc+k8+8) at spatial pg.
    // All control values are scalar (SGPR) -> s_cbranch, saddr-form loads.
    auto build_y = [&](int kc) {
        const int c0  = kc + k8;
        const int g   = c0 >> 6;
        const int sub = (c0 >> 4) & 3;
        int dlo, dhi;
        if (g == 0 || sub >= 2) { dlo = 0;  dhi = 0;  }
        else if (sub == 0)      { dlo = -g; dhi = -1; }
        else                    { dlo = 1;  dhi = g;  }
        const int lo = max(0, tt + dlo);
        const int hi = min(kT - 1, tt + dhi);

        float yv[8];
        if (lo <= hi) {                      // scalar guard; first tap = assign
            const float* xp0 =
                x + ((size_t)((nb * kT + lo) * kC + c0) * kHW + pg);
            #pragma unroll
            for (int i = 0; i < 8; ++i) yv[i] = xp0[(size_t)i * kHW];
            #pragma unroll
            for (int j = 1; j < 3; ++j) {    // up to 2 more taps
                if (lo + j <= hi) {
                    const float* xp =
                        x + ((size_t)((nb * kT + lo + j) * kC + c0) * kHW + pg);
                    #pragma unroll
                    for (int i = 0; i < 8; ++i) yv[i] += xp[(size_t)i * kHW];
                }
            }
        } else {
            #pragma unroll
            for (int i = 0; i < 8; ++i) yv[i] = 0.0f;
        }
        v8bf pk;
        #pragma unroll
        for (int i = 0; i < 8; ++i) pk[i] = (bf16_t)yv[i];
        return pk;
    };

    v8f acc[2][4];
    #pragma unroll
    for (int mi = 0; mi < 2; ++mi)
        #pragma unroll
        for (int ni = 0; ni < 4; ++ni)
            acc[mi][ni] = v8f{};

    const int row = lane & 15;
    const int kha = (lane >> 4) << 3;       // A half: K base 0 or 8
    const int khb = (lane >> 4) << 3;       // B half: K base 0 or 8

    // Prologue: fill Y buffer 0 for K-step 0.
    {
        v8bf p = build_y(0);
        *(v8bf*)&lds[kWElems + plocal * kYStride + k8] = p;
    }
    __syncthreads();

    #pragma unroll 2
    for (int s = 0; s < kKSteps; ++s) {
        const int kc = s * kKStep;
        const int ycur = kWElems + (s & 1) * kYElems;        // folds w/ unroll 2
        const int ynxt = kWElems + ((s + 1) & 1) * kYElems;

        // Pipelined: pull next K-step's taps from global before this step's MMAs.
        v8bf nxt = {};
        if (s < kKSteps - 1) nxt = build_y(kc + kKStep);

        // ---- A fragments (ISA 16-bit 16x32 layout: K {kha..+7, kha+16..+23}) ----
        v16bf afrag[2];
        #pragma unroll
        for (int mi = 0; mi < 2; ++mi) {
            const int o = (waveid * 2 + mi) * 16 + row;
            const int ai = o * kC + kc + kha;
            v8bf lo8 = *(const v8bf*)&lds[ai];
            v8bf hi8 = *(const v8bf*)&lds[ai + 16];
            #pragma unroll
            for (int e = 0; e < 8; ++e) { afrag[mi][e] = lo8[e]; afrag[mi][8 + e] = hi8[e]; }
        }

        // ---- B fragments (ISA 16-bit 32x16 layout: K {khb..+7, khb+16..+23}) ----
        v16bf bfrag[4];
        #pragma unroll
        for (int ni = 0; ni < 4; ++ni) {
            const int n  = ni * 16 + row;
            const int bi = ycur + n * kYStride + khb;
            v8bf lo8 = *(const v8bf*)&lds[bi];
            v8bf hi8 = *(const v8bf*)&lds[bi + 16];
            #pragma unroll
            for (int e = 0; e < 8; ++e) { bfrag[ni][e] = lo8[e]; bfrag[ni][8 + e] = hi8[e]; }
        }

        // ---- 8 WMMAs per wave per K-step ----
        #pragma unroll
        for (int mi = 0; mi < 2; ++mi)
            #pragma unroll
            for (int ni = 0; ni < 4; ++ni)
                acc[mi][ni] = __builtin_amdgcn_wmma_f32_16x16x32_bf16(
                    false, afrag[mi], false, bfrag[ni],
                    (short)0, acc[mi][ni], false, false);

        // Publish next tile into the alternate buffer; one barrier per step.
        if (s < kKSteps - 1)
            *(v8bf*)&lds[ynxt + plocal * kYStride + k8] = nxt;
        __syncthreads();
    }

    // ---- Write 256x64 f32 outputs, non-temporal (keep x hot in 192MB L2) ----
    // C/D layout: VGPR r -> M = r (+8 for upper half-wave), N = lane&15.
    const int mrow = (lane >> 4) << 3;
    #pragma unroll
    for (int mi = 0; mi < 2; ++mi) {
        const int obase = (waveid * 2 + mi) * 16 + mrow;
        #pragma unroll
        for (int ni = 0; ni < 4; ++ni) {
            const int pgl = p0 + ni * 16 + (lane & 15);
            #pragma unroll
            for (int r = 0; r < 8; ++r)
                __builtin_nontemporal_store(
                    acc[mi][ni][r],
                    &out[(size_t)(img * kC + obase + r) * kHW + pgl]);
        }
    }
}

extern "C" void kernel_launch(void* const* d_in, const int* in_sizes, int n_in,
                              void* d_out, int out_size, void* d_ws, size_t ws_size,
                              hipStream_t stream) {
    (void)in_sizes; (void)n_in; (void)out_size; (void)ws_size;
    const float* x    = (const float*)d_in[0];   // (64,256,56,56) f32
    const float* netw = (const float*)d_in[5];   // (256,256,1,1) f32
    float*       out  = (float*)d_out;           // (64,256,56,56) f32
    bf16_t*      wb   = (bf16_t*)d_ws;           // 128 KB bf16 weight scratch

    msconv_wcvt_kernel<<<(kC * kC + 255) / 256, 256, 0, stream>>>(netw, wb);

    dim3 grid(kHW / kNTile, kNT);  // (49, 64)
    msconv_gemm_kernel<<<grid, 256, kLdsBytes, stream>>>(x, wb, out);
}
